// DAGRN_73589969649940
// MI455X (gfx1250) — matrix operand order repaired and checked by reading, not compiled, not run on hardware
//
#include <hip/hip_runtime.h>
#include <hip/hip_bf16.h>
#include <math.h>

// Problem constants (from reference)
#define NN 50000
#define EE 800000
#define FF 64
#define HH 256
#define IDD 32
#define DYNF 16
#define BB 256
#define TT 128
#define GSTEPS 3
#define DIN 304  // HH + IDD + DYNF

typedef __bf16 bf16_t;
typedef __attribute__((ext_vector_type(16))) __bf16 v16bf;
typedef __attribute__((ext_vector_type(8)))  __bf16 v8bf;
typedef __attribute__((ext_vector_type(8)))  float  v8f;
typedef __attribute__((ext_vector_type(4)))  float  v4f;

__device__ __forceinline__ float sigf(float x){ return 1.f/(1.f+__expf(-x)); }

template<int ACT>
__device__ __forceinline__ float actf(float x){
  if constexpr (ACT==1) return tanhf(x);
  else if constexpr (ACT==2) return sigf(x);
  else if constexpr (ACT==3) return 0.5f*x*(1.f+erff(x*0.70710678118654752f)); // exact GELU
  else return x;
}

// ---------------------------------------------------------------------------
// WMMA GEMM with LDS tile staging: C[M,Nc] = act(A @ W + bias)
//   A row-major, split along K into A0 (K0 cols) + A1 (K1 cols). Call sites
//   with K1>0 guarantee K0 % 64 == 0 and K % 64 == 0, so every 64-wide
//   K-block lies in one source and tails only occur in single-source calls.
//   W row-major [K, Nc], Nc multiple of 64. M arbitrary, K arbitrary.
// Block = 256 threads = 8 waves (2x4) -> 32x64 output tile, K-step 64.
// LDS: As[32 rows][64 k] bf16 (4KB), Bs[64 cols][64 k] bf16 transposed (8KB).
// The fast/tail staging choice is a block-uniform scalar branch, so the hot
// path has only unconditional b128 loads + cvt + ds stores; 2 wmma per
// barrier pair.
// Fragment layouts per CDNA5 ISA 7.12.2 (wave32):
//   A 16x32: lanes 0-15 row=lane, K {0..7,16..23}; lanes 16-31 K {8..15,24..31}
//   B 32x16: lanes 0-15 col=lane, K {0..7,16..23}; lanes 16-31 K {8..15,24..31}
//   C 16x16 f32: VGPR j -> row = j + (lane>=16 ? 8:0), col = lane&15
// ---------------------------------------------------------------------------
template<int ACT>
__global__ void __launch_bounds__(256) k_gemm(
    const float* __restrict__ A0, int K0,
    const float* __restrict__ A1, int K1,
    const float* __restrict__ W, const float* __restrict__ bias,
    float* __restrict__ C, int M, int Nc)
{
  __shared__ unsigned short As[32*64];   // bf16 [row][k]
  __shared__ unsigned short Bs[64*64];   // bf16 [col][k] (transposed)

  const int K    = K0 + K1;
  const int tid  = threadIdx.x;
  const int lane = tid & 31;
  const int wid  = tid >> 5;
  const int rowt = blockIdx.x * 32;
  const int colt = blockIdx.y * 64;

  // cooperative A-tile load mapping: 32 rows x 64 k, 8 floats/thread
  const int arow = tid >> 3;
  const int acol = (tid & 7) * 8;
  int agr = rowt + arow; if (agr > M-1) agr = M-1;   // clamp (stores are guarded)

  // cooperative B-tile load mapping: 64 cols x 64 k, 16 k's/thread (coalesced in n)
  const int bcol = tid & 63;
  const int bk16 = (tid >> 6) * 16;
  const int bgc  = colt + bcol;

  // fragment read mapping
  const int rowl  = (wid >> 2) * 16 + (lane & 15);
  const int coll  = (wid & 3) * 16 + (lane & 15);
  const int abase = (lane >> 4) * 8;
  const int bbase = (lane >> 4) * 8;

  v8f acc = {0.f,0.f,0.f,0.f,0.f,0.f,0.f,0.f};

  for (int kb = 0; kb < K; kb += 64) {
    const int kv = (K - kb < 64) ? (K - kb) : 64;
    // uniform source select for this K-block (scalar)
    const float* __restrict__ Ap; int lda, kloc;
    if (kb < K0){ Ap = A0; lda = K0; kloc = kb; }
    else        { Ap = A1; lda = K1; kloc = kb - K0; }
    const float* pa = Ap + (size_t)agr * lda + kloc;

    if (kv == 64) {
      // ---- fast path: unconditional vector staging ----
      v4f va0 = *(const v4f*)(pa + acol);
      v4f va1 = *(const v4f*)(pa + acol + 4);
      v8bf a8;
#pragma unroll
      for (int j = 0; j < 4; ++j){ a8[j] = (bf16_t)va0[j]; a8[4+j] = (bf16_t)va1[j]; }
      *(v8bf*)&As[arow*64 + acol] = a8;

      float bvv[16];
#pragma unroll
      for (int j = 0; j < 16; ++j) bvv[j] = W[(size_t)(kb + bk16 + j) * Nc + bgc];
      v8bf b0, b1;
#pragma unroll
      for (int j = 0; j < 8; ++j){ b0[j] = (bf16_t)bvv[j]; b1[j] = (bf16_t)bvv[8+j]; }
      *(v8bf*)&Bs[bcol*64 + bk16]     = b0;
      *(v8bf*)&Bs[bcol*64 + bk16 + 8] = b1;

      // prefetch next B tile rows while this block computes
      if (kb + 64 < K)
        __builtin_prefetch(&W[(size_t)(kb + 64 + bk16) * Nc + bgc], 0, 1);
    } else {
      // ---- K-tail: guarded, zero-filled staging (runs at most once) ----
      v8bf a8;
#pragma unroll
      for (int j = 0; j < 8; ++j)
        a8[j] = (bf16_t)((acol + j < kv) ? pa[acol + j] : 0.f);
      *(v8bf*)&As[arow*64 + acol] = a8;

      v8bf b0, b1;
#pragma unroll
      for (int j = 0; j < 8; ++j){
        b0[j] = (bf16_t)((bk16 + j     < kv) ? W[(size_t)(kb + bk16 + j)     * Nc + bgc] : 0.f);
        b1[j] = (bf16_t)((bk16 + j + 8 < kv) ? W[(size_t)(kb + bk16 + j + 8) * Nc + bgc] : 0.f);
      }
      *(v8bf*)&Bs[bcol*64 + bk16]     = b0;
      *(v8bf*)&Bs[bcol*64 + bk16 + 8] = b1;
    }

    __syncthreads();

    // ---- 2 wmma per staged 64-K block ----
#pragma unroll
    for (int c = 0; c < 2; ++c) {
      union { v16bf v; v8bf h[2]; } au, bu;
      au.h[0] = *(const v8bf*)&As[rowl*64 + c*32 + abase];
      au.h[1] = *(const v8bf*)&As[rowl*64 + c*32 + abase + 16];
      bu.h[0] = *(const v8bf*)&Bs[coll*64 + c*32 + bbase];
      bu.h[1] = *(const v8bf*)&Bs[coll*64 + c*32 + bbase + 16];
      acc = __builtin_amdgcn_wmma_f32_16x16x32_bf16(
          /*neg_a=*/false, au.v, /*neg_b=*/false, bu.v,
          /*c_mod=*/(short)0, acc, /*reuse_a=*/false, /*reuse_b=*/false);
    }

    __syncthreads();
  }

  const int   crow0 = rowt + (wid >> 2) * 16 + (lane >> 4) * 8;
  const int   ccol  = colt + (wid & 3) * 16 + (lane & 15);
  const float bv    = bias[ccol];
#pragma unroll
  for (int j = 0; j < 8; ++j) {
    int r = crow0 + j;
    if (r < M) C[(size_t)r * Nc + ccol] = actf<ACT>(acc[j] + bv);
  }
}

// ---------------------------------------------------------------------------
// Elementwise / gather / scatter kernels
// ---------------------------------------------------------------------------
__global__ void k_zero(float* __restrict__ p, size_t n){
  size_t i = (size_t)blockIdx.x * blockDim.x + threadIdx.x;
  if (i < n) p[i] = 0.f;
}

// h_agg[dst] += h[src] ; thread handles 4 consecutive channels of one edge
__global__ void k_segsum(const float* __restrict__ h, const int* __restrict__ src,
                         const int* __restrict__ dst, float* __restrict__ hagg){
  size_t i = (size_t)blockIdx.x * blockDim.x + threadIdx.x;
  if (i >= (size_t)EE * (HH/4)) return;
  int e  = (int)(i >> 6);            // HH/4 = 64 chunks per edge
  int c0 = (int)(i & 63) * 4;
  int s = src[e], d = dst[e];
  const float* hp = h    + (size_t)s * HH + c0;
  float*       ap = hagg + (size_t)d * HH + c0;
  v4f v = *(const v4f*)hp;
  atomicAdd(ap + 0, v[0]);
  atomicAdd(ap + 1, v[1]);
  atomicAdd(ap + 2, v[2]);
  atomicAdd(ap + 3, v[3]);
}

__global__ void k_mul_inplace(float* __restrict__ a, const float* __restrict__ b, size_t n){
  size_t i = (size_t)blockIdx.x * blockDim.x + threadIdx.x;
  if (i < n) a[i] *= b[i];
}

// h = (1-zg)*hagg + zg*ht
__global__ void k_hupd(float* __restrict__ h, const float* __restrict__ zg,
                       const float* __restrict__ hagg, const float* __restrict__ ht, size_t n){
  size_t i = (size_t)blockIdx.x * blockDim.x + threadIdx.x;
  if (i < n){ float z = zg[i]; h[i] = (1.f - z)*hagg[i] + z*ht[i]; }
}

// rnn_in[b,t,:] = [ h[traj]*(1+g)+beta , id_emb[traj] , dyn[b] ]
__global__ void k_build_rnnin(const float* __restrict__ h, const float* __restrict__ film,
                              const float* __restrict__ id_emb, const float* __restrict__ dyn,
                              const int* __restrict__ traj, float* __restrict__ rnnin){
  size_t i = (size_t)blockIdx.x * blockDim.x + threadIdx.x;
  if (i >= (size_t)BB * TT * DIN) return;
  int    c  = (int)(i % DIN);
  size_t bt = i / DIN;
  int    b  = (int)(bt / TT);
  int node  = traj[bt];
  float v;
  if (c < HH)
    v = h[(size_t)node * HH + c] * (1.f + film[(size_t)b * 2*HH + c]) + film[(size_t)b * 2*HH + HH + c];
  else if (c < HH + IDD)
    v = id_emb[(size_t)node * IDD + (c - HH)];
  else
    v = dyn[(size_t)b * DYNF + (c - HH - IDD)];
  rnnin[i] = v;
}

// one GRU timestep gate math for both directions (dir = idx>>16)
__global__ void k_gru_gate(const float* __restrict__ xgf, const float* __restrict__ xgb,
                           const float* __restrict__ ghf, const float* __restrict__ ghb,
                           float* __restrict__ hf, float* __restrict__ hb,
                           float* __restrict__ outf, float* __restrict__ outrb,
                           const int* __restrict__ lengths, int t){
  int idx = blockIdx.x * blockDim.x + threadIdx.x;
  if (idx >= 2 * BB * HH) return;
  int dir = idx >> 16;        // BB*HH == 65536
  int rem = idx & 65535;
  int b   = rem >> 8;         // HH == 256
  int c   = rem & 255;
  int len = lengths[b];
  float m = (t < len) ? 1.f : 0.f;
  const float *xg, *gh; float *h, *o; int xrow;
  if (dir == 0){ xg = xgf; gh = ghf; h = hf; o = outf;  xrow = b*TT + t; }
  else { int rv = len - 1 - t; if (rv < 0) rv = 0;
         xg = xgb; gh = ghb; h = hb; o = outrb; xrow = b*TT + rv; }
  size_t xo = (size_t)xrow * (3*HH);
  float xr = xg[xo + c], xz = xg[xo + HH + c], xn = xg[xo + 2*HH + c];
  size_t go = (size_t)b * (3*HH);
  float hr = gh[go + c], hz = gh[go + HH + c], hn = gh[go + 2*HH + c];
  float hv = h[b*HH + c];
  float r  = sigf(xr + hr);
  float z  = sigf(xz + hz);
  float nn = tanhf(xn + r*hn);
  float hnew = (1.f - z)*nn + z*hv;
  float hout = (m > 0.f) ? hnew : hv;
  h[b*HH + c] = hout;
  o[((size_t)b*TT + t)*HH + c] = hout * m;
}

// out_b[b,t] = out_rb[b, max(len-1-t,0)] * mask
__global__ void k_unrev(const float* __restrict__ outrb, const int* __restrict__ lengths,
                        float* __restrict__ outb){
  size_t i = (size_t)blockIdx.x * blockDim.x + threadIdx.x;
  if (i >= (size_t)BB * TT * HH) return;
  int    c  = (int)(i % HH);
  size_t bt = i / HH;
  int    t  = (int)(bt % TT);
  int    b  = (int)(bt / TT);
  int len = lengths[b];
  if (t < len){
    int rv = len - 1 - t; if (rv < 0) rv = 0;
    outb[i] = outrb[((size_t)b*TT + rv)*HH + c];
  } else outb[i] = 0.f;
}

// score_in = [out_f | out_b | dyn]
__global__ void k_build_scorein(const float* __restrict__ outf, const float* __restrict__ outb,
                                const float* __restrict__ dyn, float* __restrict__ scin){
  const int W = 2*HH + DYNF; // 528
  size_t i = (size_t)blockIdx.x * blockDim.x + threadIdx.x;
  if (i >= (size_t)BB * TT * W) return;
  int    c  = (int)(i % W);
  size_t bt = i / W;
  int    b  = (int)(bt / TT);
  float v;
  if (c < HH)        v = outf[bt*HH + c];
  else if (c < 2*HH) v = outb[bt*HH + (c - HH)];
  else               v = dyn[(size_t)b * DYNF + (c - 2*HH)];
  scin[i] = v;
}

// scores[row] = dot(hattn[row], attn_W2) + b2, masked where traj==0
__global__ void __launch_bounds__(256) k_scores(const float* __restrict__ hattn,
                                                const float* __restrict__ W2, const float* __restrict__ b2,
                                                const int* __restrict__ traj, float* __restrict__ scores){
  int lane = threadIdx.x & 31;
  int row  = (blockIdx.x * blockDim.x + threadIdx.x) >> 5;
  if (row >= BB*TT) return;
  float s = 0.f;
  for (int c = lane; c < HH; c += 32) s += hattn[(size_t)row*HH + c] * W2[c];
  for (int off = 16; off; off >>= 1) s += __shfl_xor(s, off);
  if (lane == 0){
    s += b2[0];
    if (traj[row] == 0) s = -1e9f;
    scores[row] = s;
  }
}

// softmax over T + context = sum_t alpha * [out_f|out_b]; one block per batch row
__global__ void __launch_bounds__(128) k_attn_ctx(const float* __restrict__ scores,
                                                  const float* __restrict__ outf, const float* __restrict__ outb,
                                                  float* __restrict__ ctx){
  __shared__ float red[TT];
  __shared__ float alpha[TT];
  int b = blockIdx.x, t = threadIdx.x;
  float s = scores[(size_t)b*TT + t];
  red[t] = s; __syncthreads();
  for (int off = TT/2; off; off >>= 1){ if (t < off) red[t] = fmaxf(red[t], red[t+off]); __syncthreads(); }
  float mx = red[0]; __syncthreads();
  float e = __expf(s - mx);
  red[t] = e; __syncthreads();
  for (int off = TT/2; off; off >>= 1){ if (t < off) red[t] += red[t+off]; __syncthreads(); }
  float inv = 1.f / red[0]; __syncthreads();
  alpha[t] = e * inv; __syncthreads();
  for (int c = t; c < 2*HH; c += TT){
    float a = 0.f;
    for (int t2 = 0; t2 < TT; ++t2){
      float v = (c < HH) ? outf[((size_t)b*TT + t2)*HH + c]
                         : outb[((size_t)b*TT + t2)*HH + (c - HH)];
      a += alpha[t2] * v;
    }
    ctx[(size_t)b*2*HH + c] = a;
  }
}

// rnn_last = LayerNorm([h_f, h_b]) * g + beta ; one wave per row
__global__ void __launch_bounds__(256) k_lnorm(const float* __restrict__ hf, const float* __restrict__ hb,
                                               const float* __restrict__ g, const float* __restrict__ bta,
                                               float* __restrict__ outp){
  int lane = threadIdx.x & 31;
  int row  = (blockIdx.x * blockDim.x + threadIdx.x) >> 5;
  if (row >= BB) return;
  float vals[16]; float sum = 0.f, sq = 0.f;
#pragma unroll
  for (int i = 0; i < 16; ++i){
    int c = lane + 32*i;
    float v = (c < HH) ? hf[(size_t)row*HH + c] : hb[(size_t)row*HH + (c - HH)];
    vals[i] = v; sum += v; sq += v*v;
  }
  for (int off = 16; off; off >>= 1){ sum += __shfl_xor(sum, off); sq += __shfl_xor(sq, off); }
  float mu = sum / (2.f*HH);
  float var = sq / (2.f*HH) - mu*mu;
  float rs = rsqrtf(var + 1e-5f);
#pragma unroll
  for (int i = 0; i < 16; ++i){
    int c = lane + 32*i;
    outp[(size_t)row*2*HH + c] = (vals[i] - mu)*rs*g[c] + bta[c];
  }
}

// gate_in = [context, rnn_last, dyn]
__global__ void k_build_gatein(const float* __restrict__ ctx, const float* __restrict__ rl,
                               const float* __restrict__ dyn, float* __restrict__ gi){
  const int W = 4*HH + DYNF; // 1040
  size_t i = (size_t)blockIdx.x * blockDim.x + threadIdx.x;
  if (i >= (size_t)BB * W) return;
  int c = (int)(i % W); int b = (int)(i / W);
  float v;
  if (c < 2*HH)        v = ctx[(size_t)b*2*HH + c];
  else if (c < 4*HH)   v = rl[(size_t)b*2*HH + (c - 2*HH)];
  else                 v = dyn[(size_t)b*DYNF + (c - 4*HH)];
  gi[i] = v;
}

// final = zf*ctx + (1-zf)*rnn_last ; fcin = [final, dyn]
__global__ void k_final_fcin(const float* __restrict__ zf, const float* __restrict__ ctx,
                             const float* __restrict__ rl, const float* __restrict__ dyn,
                             float* __restrict__ fcin){
  const int W = 2*HH + DYNF; // 528
  size_t i = (size_t)blockIdx.x * blockDim.x + threadIdx.x;
  if (i >= (size_t)BB * W) return;
  int c = (int)(i % W); int b = (int)(i / W);
  float v;
  if (c < 2*HH){
    float z = zf[(size_t)b*2*HH + c];
    v = z*ctx[(size_t)b*2*HH + c] + (1.f - z)*rl[(size_t)b*2*HH + c];
  } else v = dyn[(size_t)b*DYNF + (c - 2*HH)];
  fcin[i] = v;
}

// out[b] = dot(hfc[b], fc_W2) + fc_b2 ; one wave per row
__global__ void __launch_bounds__(256) k_out(const float* __restrict__ hfc,
                                             const float* __restrict__ W2, const float* __restrict__ b2,
                                             float* __restrict__ outp){
  int lane = threadIdx.x & 31;
  int row  = (blockIdx.x * blockDim.x + threadIdx.x) >> 5;
  if (row >= BB) return;
  float s = 0.f;
  for (int c = lane; c < HH; c += 32) s += hfc[(size_t)row*HH + c] * W2[c];
  for (int off = 16; off; off >>= 1) s += __shfl_xor(s, off);
  if (lane == 0) outp[row] = s + b2[0];
}

// ---------------------------------------------------------------------------
// Host side
// ---------------------------------------------------------------------------
static void gemm(hipStream_t s, int act, const float* A0, int K0, const float* A1, int K1,
                 const float* W, const float* bias, float* C, int M, int Nc){
  if (!A1){ A1 = A0; K1 = 0; }
  dim3 g((unsigned)((M + 31)/32), (unsigned)(Nc/64)), b(256);
  switch (act){
    case 1: k_gemm<1><<<g,b,0,s>>>(A0,K0,A1,K1,W,bias,C,M,Nc); break;
    case 2: k_gemm<2><<<g,b,0,s>>>(A0,K0,A1,K1,W,bias,C,M,Nc); break;
    case 3: k_gemm<3><<<g,b,0,s>>>(A0,K0,A1,K1,W,bias,C,M,Nc); break;
    default: k_gemm<0><<<g,b,0,s>>>(A0,K0,A1,K1,W,bias,C,M,Nc); break;
  }
}
static inline unsigned nblk(size_t n){ return (unsigned)((n + 255)/256); }

extern "C" void kernel_launch(void* const* d_in, const int* in_sizes, int n_in,
                              void* d_out, int out_size, void* d_ws, size_t ws_size,
                              hipStream_t stream)
{
  (void)in_sizes; (void)n_in; (void)out_size; (void)ws_size;
  const float* x0    = (const float*)d_in[0];
  const float* dyn   = (const float*)d_in[1];
  const float* Wproj = (const float*)d_in[2];
  const float* bproj = (const float*)d_in[3];
  const float* Wz    = (const float*)d_in[4];
  const float* bz    = (const float*)d_in[5];
  const float* Wr    = (const float*)d_in[6];
  const float* br    = (const float*)d_in[7];
  const float* Wn    = (const float*)d_in[8];
  const float* bnn   = (const float*)d_in[9];
  const float* id_emb= (const float*)d_in[10];
  const float* fW1   = (const float*)d_in[11];
  const float* fb1   = (const float*)d_in[12];
  const float* fW2   = (const float*)d_in[13];
  const float* fb2   = (const float*)d_in[14];
  const float* Wi_f  = (const float*)d_in[15];
  const float* Wh_f  = (const float*)d_in[16];
  const float* bi_f  = (const float*)d_in[17];
  const float* bh_f  = (const float*)d_in[18];
  const float* Wi_b  = (const float*)d_in[19];
  const float* Wh_b  = (const float*)d_in[20];
  const float* bi_b  = (const float*)d_in[21];
  const float* bh_b  = (const float*)d_in[22];
  const float* ln_g  = (const float*)d_in[23];
  const float* ln_b  = (const float*)d_in[24];
  const float* aW1   = (const float*)d_in[25];
  const float* ab1   = (const float*)d_in[26];
  const float* aW2   = (const float*)d_in[27];
  const float* ab2   = (const float*)d_in[28];
  const float* gW    = (const float*)d_in[29];
  const float* gb    = (const float*)d_in[30];
  const float* cW1   = (const float*)d_in[31];
  const float* cb1   = (const float*)d_in[32];
  const float* cW2   = (const float*)d_in[33];
  const float* cb2   = (const float*)d_in[34];
  const int*   traj  = (const int*)d_in[35];
  const int*   lens  = (const int*)d_in[36];
  const int*   esrc  = (const int*)d_in[37];
  const int*   edst  = (const int*)d_in[38];
  float* out = (float*)d_out;
  float* ws  = (float*)d_ws;

  // --- workspace layout (floats). Phase-overlapped; needs ~370 MB. ---
  const size_t NH = (size_t)NN * HH;                 // 12.8M
  float* h    = ws;                                  // persists until rnn_in built
  float* gnn  = ws + NH;                             // 4*NH region, multi-use
  float* hagg = gnn;
  float* zg   = gnn + NH;
  float* rg   = gnn + 2*NH;
  float* ht   = gnn + 3*NH;
  float* xgf  = gnn;                                 // reuse after GNN: B*T*768
  float* xgb  = gnn + (size_t)BB*TT*3*HH;
  float* scin = gnn;                                 // reuse after scan: B*T*528
  float* hattn= gnn + (size_t)BB*TT*(2*HH+DYNF);
  size_t m = NH + 4*NH;
  float* rnnin = ws + m;                             // B*T*304
  float* outf  = rnnin;                              // reuse after xg GEMMs
  m += (size_t)BB*TT*DIN;
  float* outrb = ws + m; m += (size_t)BB*TT*HH;
  float* outb  = ws + m; m += (size_t)BB*TT*HH;
  float* filmh = ws + m; m += (size_t)BB*HH;
  float* film  = ws + m; m += (size_t)BB*2*HH;
  float* hf    = ws + m; m += (size_t)BB*HH;
  float* hb    = ws + m; m += (size_t)BB*HH;
  float* ghf   = ws + m; m += (size_t)BB*3*HH;
  float* ghb   = ws + m; m += (size_t)BB*3*HH;
  float* scores= ws + m; m += (size_t)BB*TT;
  float* ctx   = ws + m; m += (size_t)BB*2*HH;
  float* rlast = ws + m; m += (size_t)BB*2*HH;
  float* gatein= ws + m; m += (size_t)BB*(4*HH+DYNF);
  float* zf    = ws + m; m += (size_t)BB*2*HH;
  float* fcin  = ws + m; m += (size_t)BB*(2*HH+DYNF);
  float* hfc   = ws + m; m += (size_t)BB*HH;

  // ---- GNN phase ----
  gemm(stream, 1, x0, FF, nullptr, 0, Wproj, bproj, h, NN, HH);   // h = tanh(x0@Wproj+b)
  for (int s = 0; s < GSTEPS; ++s){
    k_zero<<<nblk(NH), 256, 0, stream>>>(hagg, NH);
    k_segsum<<<nblk((size_t)EE*(HH/4)), 256, 0, stream>>>(h, esrc, edst, hagg);
    gemm(stream, 2, x0, FF, hagg, HH, Wz, bz, zg, NN, HH);        // sigmoid
    gemm(stream, 2, x0, FF, hagg, HH, Wr, br, rg, NN, HH);        // sigmoid
    k_mul_inplace<<<nblk(NH), 256, 0, stream>>>(rg, hagg, NH);    // rg *= h_agg
    gemm(stream, 1, x0, FF, rg, HH, Wn, bnn, ht, NN, HH);         // tanh
    k_hupd<<<nblk(NH), 256, 0, stream>>>(h, zg, hagg, ht, NH);
  }

  // ---- FiLM + sequence input ----
  gemm(stream, 0, dyn, DYNF, nullptr, 0, fW1, fb1, filmh, BB, HH);
  gemm(stream, 1, filmh, HH, nullptr, 0, fW2, fb2, film, BB, 2*HH);
  k_build_rnnin<<<nblk((size_t)BB*TT*DIN), 256, 0, stream>>>(h, film, id_emb, dyn, traj, rnnin);

  // GRU input projections (time-parallel)
  gemm(stream, 0, rnnin, DIN, nullptr, 0, Wi_f, bi_f, xgf, BB*TT, 3*HH);
  gemm(stream, 0, rnnin, DIN, nullptr, 0, Wi_b, bi_b, xgb, BB*TT, 3*HH);

  // ---- bidirectional GRU scan ----
  k_zero<<<nblk((size_t)BB*HH), 256, 0, stream>>>(hf, (size_t)BB*HH);
  k_zero<<<nblk((size_t)BB*HH), 256, 0, stream>>>(hb, (size_t)BB*HH);
  for (int t = 0; t < TT; ++t){
    gemm(stream, 0, hf, HH, nullptr, 0, Wh_f, bh_f, ghf, BB, 3*HH);
    gemm(stream, 0, hb, HH, nullptr, 0, Wh_b, bh_b, ghb, BB, 3*HH);
    k_gru_gate<<<nblk((size_t)2*BB*HH), 256, 0, stream>>>(xgf, xgb, ghf, ghb, hf, hb, outf, outrb, lens, t);
  }
  k_unrev<<<nblk((size_t)BB*TT*HH), 256, 0, stream>>>(outrb, lens, outb);

  // ---- attention ----
  k_build_scorein<<<nblk((size_t)BB*TT*(2*HH+DYNF)), 256, 0, stream>>>(outf, outb, dyn, scin);
  gemm(stream, 1, scin, 2*HH+DYNF, nullptr, 0, aW1, ab1, hattn, BB*TT, HH);
  k_scores<<<nblk((size_t)BB*TT*32), 256, 0, stream>>>(hattn, aW2, ab2, traj, scores);
  k_attn_ctx<<<BB, TT, 0, stream>>>(scores, outf, outb, ctx);

  // ---- head ----
  k_lnorm<<<nblk((size_t)BB*32), 256, 0, stream>>>(hf, hb, ln_g, ln_b, rlast);
  k_build_gatein<<<nblk((size_t)BB*(4*HH+DYNF)), 256, 0, stream>>>(ctx, rlast, dyn, gatein);
  gemm(stream, 2, gatein, 4*HH+DYNF, nullptr, 0, gW, gb, zf, BB, 2*HH);
  k_final_fcin<<<nblk((size_t)BB*(2*HH+DYNF)), 256, 0, stream>>>(zf, ctx, rlast, dyn, fcin);
  gemm(stream, 3, fcin, 2*HH+DYNF, nullptr, 0, cW1, cb1, hfc, BB, HH);   // exact GELU
  k_out<<<nblk((size_t)BB*32), 256, 0, stream>>>(hfc, cW2, cb2, out);
}
